// TransformerEncoderLayer_29240137351737
// MI455X (gfx1250) — matrix operand order repaired and checked by reading, not compiled, not run on hardware
//
#include <hip/hip_runtime.h>
#include <hip/hip_bf16.h>

// ---------------------------------------------------------------------------
// Transformer encoder layer for MI455X (gfx1250, wave32, WMMA).
// All matmuls use v_wmma_f32_16x16x32_bf16 (f32 accumulate).
// A-fragment loads are ds_load_b128 pairs (B: strided 16-bit loads); attention K/V tiles use
// global_load_async_to_lds_b128 (ASYNCcnt path).
// ---------------------------------------------------------------------------

typedef __bf16 bf16;
typedef __attribute__((ext_vector_type(16))) __bf16 v16bf;
typedef __attribute__((ext_vector_type(8)))  float  v8f;

#define DEV static __device__ __forceinline__

// Problem constants (fixed by the reference).
#define BN 4
#define SN 2048
#define DN 768
#define HN 12
#define DKN 64
#define FN 3072
#define MN (BN * SN)   // 8192 token rows

// Async copy: 16 bytes global -> LDS per lane (GLOBAL_LOAD_ASYNC_TO_LDS_B128,
// GV mode: VDST = LDS byte address, VADDR = 64-bit global address).
#define ASYNC_LDS_B128(ldsp, gp)                                              \
    asm volatile("global_load_async_to_lds_b128 %0, %1, off"                  \
                 :: "v"((unsigned)(uintptr_t)(ldsp)),                         \
                    "v"((unsigned long long)(uintptr_t)(gp))                  \
                 : "memory")
#define WAIT_ASYNC() asm volatile("s_wait_asynccnt 0" ::: "memory")

DEV bf16 f2b(float f) {
    unsigned u = __builtin_bit_cast(unsigned, f);
    u += 0x7FFFu + ((u >> 16) & 1u);           // round-to-nearest-even
    unsigned short h = (unsigned short)(u >> 16);
    return __builtin_bit_cast(bf16, h);
}

struct U32x8 { uint4 a, b; };
DEV v16bf as_v16bf(uint4 a, uint4 b) {
    U32x8 t{a, b};
    return __builtin_bit_cast(v16bf, t);
}

DEV v8f wmma_bf16(v16bf a, v16bf b, v8f c) {
    return __builtin_amdgcn_wmma_f32_16x16x32_bf16(
        false, a, false, b, (short)0, c, false, false);
}

// A fragment (16x32 bf16, ISA 7.12.2): lane r of each half owns row m0+r;
// its 16 elements are TWO contiguous 8-element runs of that row:
//   elements 0..7  = k0 + 8*half .. +7
//   elements 8..15 = k0 + 16 + 8*half .. +7
// => two ds_load_b128.
DEV v16bf loadA(const bf16* base, int ld, int m0, int k0, int hl, int r) {
    const bf16* p = base + (size_t)(m0 + r) * ld + k0 + 8 * hl;
    return as_v16bf(*(const uint4*)p, *(const uint4*)(p + 16));
}

// B fragment (32x16 bf16): lane holds column n0 + r of a row-major [K x N]
// tile; elements 0..7 = rows k0 + 8*half .. +7, 8..15 = k0 + 16 + 8*half .. +7.
DEV v16bf loadB_row(const bf16* base, int ld, int k0, int n0, int hl, int r) {
    const bf16* p = base + (size_t)(k0 + 8 * hl) * ld + n0 + r;
    v16bf v;
#pragma unroll
    for (int s = 0; s < 8; ++s) {
        v[s] = p[(size_t)s * ld];
        v[s + 8] = p[(size_t)(s + 16) * ld];
    }
    return v;
}

// ---------------------------------------------------------------------------
// Prep kernels
// ---------------------------------------------------------------------------

__global__ void cast_bf16_kernel(const float* __restrict__ src,
                                 bf16* __restrict__ dst, long n) {
    long i = (long)blockIdx.x * blockDim.x + threadIdx.x;
    if (i < n) dst[i] = f2b(src[i]);
}

// Wq/Wk/Wv are [H, D, DK]; build row-major bf16 B matrix [D, 3*H*DK=2304].
__global__ void pack_qkvw_kernel(const float* __restrict__ Wq,
                                 const float* __restrict__ Wk,
                                 const float* __restrict__ Wv,
                                 bf16* __restrict__ dst) {
    long i = (long)blockIdx.x * blockDim.x + threadIdx.x;
    if (i >= (long)DN * 2304) return;
    int d = (int)(i / 2304), n = (int)(i % 2304);
    const float* W = (n < 768) ? Wq : (n < 1536 ? Wk : Wv);
    int nn = n % 768;
    int h = nn >> 6, kk = nn & 63;
    dst[i] = f2b(W[(size_t)h * DN * DKN + (size_t)d * DKN + kk]);
}

__global__ void pack_qkvb_kernel(const float* __restrict__ bq,
                                 const float* __restrict__ bk,
                                 const float* __restrict__ bv,
                                 float* __restrict__ dst) {
    int i = blockIdx.x * blockDim.x + threadIdx.x;
    if (i >= 2304) return;
    dst[i] = (i < 768) ? bq[i] : (i < 1536 ? bk[i - 768] : bv[i - 1536]);
}

// ---------------------------------------------------------------------------
// Tiled bf16 GEMM: C[M,N] = A[M,K] * B[K,N] + bias, WMMA inner loop.
// 128x128 block tile, 256 threads = 8 waves (2 x 4), 64x32 per wave.
// LDS leading dims padded (40 / 136) for conflict-free fragment loads.
// MODE 0: scatter epilogue -> q [B,H,S,DK] (x0.125), kT [B,H,DK,S], v [B,H,S,DK]
// MODE 1: bf16 output with ReLU  (FFN1)
// MODE 2: f32 output             (out-proj, FFN2)
// ---------------------------------------------------------------------------
#define LDA 40    // 32 + 8 pad : 80B rows, bank stride 20 words (distinct x16)
#define LDB 136   // 128 + 8 pad: 272B rows, bank stride 68 words (distinct x16)

template <int MODE>
__global__ __launch_bounds__(256)
void gemm128_kernel(const bf16* __restrict__ A, const bf16* __restrict__ Bm,
                    const float* __restrict__ bias, int M, int N, int K,
                    float* __restrict__ outF, bf16* __restrict__ outH,
                    bf16* __restrict__ qo, bf16* __restrict__ ko,
                    bf16* __restrict__ vo) {
    __shared__ bf16 As[128 * LDA];
    __shared__ bf16 Bs[32 * LDB];

    const int tid  = threadIdx.x;
    const int lane = tid & 31, wave = tid >> 5;
    const int hl = lane >> 4, r = lane & 15;
    const int wM = wave >> 2, wN = wave & 3;      // 2 x 4 wave grid
    const int blockM = blockIdx.y * 128, blockN = blockIdx.x * 128;

    v8f acc[4][2];
#pragma unroll
    for (int i = 0; i < 4; ++i)
#pragma unroll
        for (int j = 0; j < 2; ++j) acc[i][j] = (v8f){0, 0, 0, 0, 0, 0, 0, 0};

    for (int kt = 0; kt < K; kt += 32) {
        __syncthreads();
#pragma unroll
        for (int p = 0; p < 2; ++p) {
            int e = (tid + p * 256) * 8;
            int ra = e >> 5, ca = e & 31;                 // A: 128 x 32
            *(uint4*)&As[ra * LDA + ca] =
                *(const uint4*)&A[(size_t)(blockM + ra) * K + kt + ca];
            int rb = e >> 7, cb = e & 127;                // B: 32 x 128
            *(uint4*)&Bs[rb * LDB + cb] =
                *(const uint4*)&Bm[(size_t)(kt + rb) * N + blockN + cb];
            if (kt + 32 < K) {
                __builtin_prefetch(&A[(size_t)(blockM + ra) * K + kt + 32 + ca], 0, 0);
                __builtin_prefetch(&Bm[(size_t)(kt + 32 + rb) * N + blockN + cb], 0, 0);
            }
        }
        __syncthreads();

        v16bf af[4];
#pragma unroll
        for (int i = 0; i < 4; ++i)
            af[i] = loadA(As, LDA, wM * 64 + 16 * i, 0, hl, r);
        v16bf bfr[2];
#pragma unroll
        for (int j = 0; j < 2; ++j)
            bfr[j] = loadB_row(Bs, LDB, 0, wN * 32 + 16 * j, hl, r);
#pragma unroll
        for (int i = 0; i < 4; ++i)
#pragma unroll
            for (int j = 0; j < 2; ++j)
                acc[i][j] = wmma_bf16(af[i], bfr[j], acc[i][j]);
    }

    // Epilogue. C layout: row = tile + e + 8*hl, col = tile + r.
#pragma unroll
    for (int i = 0; i < 4; ++i) {
#pragma unroll
        for (int j = 0; j < 2; ++j) {
            int n = blockN + wN * 32 + j * 16 + r;
            float bv = bias[n];
#pragma unroll
            for (int e = 0; e < 8; ++e) {
                int m = blockM + wM * 64 + i * 16 + e + 8 * hl;
                float c = acc[i][j][e] + bv;
                if (MODE == 0) {
                    int b = m >> 11, s = m & (SN - 1);    // S = 2048
                    int seg = n / 768, hh = (n % 768) >> 6, kk = n & 63;
                    if (seg == 0) {        // q: [B,H,S,DK], pre-scaled
                        qo[(((size_t)(b * HN + hh)) * SN + s) * DKN + kk] =
                            f2b(c * 0.125f);
                    } else if (seg == 1) { // k stored TRANSPOSED: [B,H,DK,S]
                        ko[(((size_t)(b * HN + hh)) * DKN + kk) * SN + s] =
                            f2b(c);
                    } else {               // v: [B,H,S,DK]
                        vo[(((size_t)(b * HN + hh)) * SN + s) * DKN + kk] =
                            f2b(c);
                    }
                } else if (MODE == 1) {
                    outH[(size_t)m * N + n] = f2b(fmaxf(c, 0.0f));
                } else {
                    outF[(size_t)m * N + n] = c;
                }
            }
        }
    }
}

// ---------------------------------------------------------------------------
// Flash attention: one block per (b*H, 64-query tile); 4 waves x 16 rows.
// q pre-scaled by 1/sqrt(DK); K comes in transposed [B,H,DK,S].
// K/V tiles stream in via global_load_async_to_lds_b128.
// ---------------------------------------------------------------------------
#define LDT 72    // 64 + 8 pad: 144B rows, bank stride 36 words (distinct x16)

__global__ __launch_bounds__(128)
void attn_kernel(const bf16* __restrict__ Q, const bf16* __restrict__ KT,
                 const bf16* __restrict__ Vg_, bf16* __restrict__ cat) {
    __shared__ bf16 Qs[64 * LDT];
    __shared__ bf16 Kts[64 * LDT];   // [dk][key]
    __shared__ bf16 Vs[64 * LDT];    // [key][dk]
    __shared__ bf16 Ps[4][16 * LDT];

    const int tid = threadIdx.x, lane = tid & 31, wave = tid >> 5;
    const int hl = lane >> 4, r = lane & 15;
    const int bh = blockIdx.y;
    const int b = bh / HN, h = bh % HN;
    const int q0 = blockIdx.x * 64;

    const bf16* Qg  = Q   + (size_t)bh * SN * DKN + (size_t)q0 * DKN; // [s][dk]
    const bf16* Ktg = KT  + (size_t)bh * DKN * SN;                    // [dk][s]
    const bf16* Vg  = Vg_ + (size_t)bh * SN * DKN;                    // [s][dk]

#pragma unroll
    for (int p = 0; p < 4; ++p) {                  // Q tile 64x64
        int e = (tid + p * 128) * 8;
        int row = e >> 6, col = e & 63;
        *(uint4*)&Qs[row * LDT + col] = *(const uint4*)&Qg[e];
    }

    float mrow[8], lrow[8];
    v8f o[4];
#pragma unroll
    for (int i = 0; i < 8; ++i) { mrow[i] = -1e30f; lrow[i] = 0.0f; }
#pragma unroll
    for (int j = 0; j < 4; ++j) o[j] = (v8f){0, 0, 0, 0, 0, 0, 0, 0};

    for (int kt = 0; kt < SN; kt += 64) {
        __syncthreads();
#pragma unroll
        for (int p = 0; p < 4; ++p) {
            int e = (tid + p * 128) * 8;
            int row = e >> 6, col = e & 63;
            // K^T tile: rows = dk, cols = keys kt..kt+63
            ASYNC_LDS_B128(&Kts[row * LDT + col],
                           &Ktg[(size_t)row * SN + kt + col]);
            // V tile: rows = keys, cols = dk
            ASYNC_LDS_B128(&Vs[row * LDT + col],
                           &Vg[(size_t)(kt + row) * DKN + col]);
        }
        WAIT_ASYNC();
        __syncthreads();

        // S = q . K^T : 16 rows x 64 keys per wave (DK=64 => 2 k-steps).
        v16bf a0 = loadA(Qs, LDT, wave * 16, 0, hl, r);
        v16bf a1 = loadA(Qs, LDT, wave * 16, 32, hl, r);
        v8f sc[4];
#pragma unroll
        for (int j = 0; j < 4; ++j) {
            sc[j] = (v8f){0, 0, 0, 0, 0, 0, 0, 0};
            sc[j] = wmma_bf16(a0, loadB_row(Kts, LDT, 0,  j * 16, hl, r), sc[j]);
            sc[j] = wmma_bf16(a1, loadB_row(Kts, LDT, 32, j * 16, hl, r), sc[j]);
        }

        // Online softmax (row = i + 8*hl spread over 16 lanes of this half).
#pragma unroll
        for (int i = 0; i < 8; ++i) {
            float rm = fmaxf(fmaxf(sc[0][i], sc[1][i]),
                             fmaxf(sc[2][i], sc[3][i]));
#pragma unroll
            for (int msk = 8; msk >= 1; msk >>= 1)
                rm = fmaxf(rm, __shfl_xor(rm, msk, 32));
            float mn = fmaxf(mrow[i], rm);
            float alpha = __expf(mrow[i] - mn);
            float rs = 0.0f;
#pragma unroll
            for (int j = 0; j < 4; ++j) {
                sc[j][i] = __expf(sc[j][i] - mn);
                rs += sc[j][i];
            }
#pragma unroll
            for (int msk = 8; msk >= 1; msk >>= 1)
                rs += __shfl_xor(rs, msk, 32);
            lrow[i] = lrow[i] * alpha + rs;
            mrow[i] = mn;
#pragma unroll
            for (int j = 0; j < 4; ++j) o[j][i] *= alpha;
        }

        // Stage P to per-wave LDS (C layout -> A layout via memory).
        bf16* Pw = Ps[wave];
#pragma unroll
        for (int j = 0; j < 4; ++j)
#pragma unroll
            for (int i = 0; i < 8; ++i)
                Pw[(i + 8 * hl) * LDT + j * 16 + r] = f2b(sc[j][i]);

        // O += P . V (64 keys => 2 k-steps, 4 n-tiles over DK=64).
        v16bf pa0 = loadA(Pw, LDT, 0, 0, hl, r);
        v16bf pa1 = loadA(Pw, LDT, 0, 32, hl, r);
#pragma unroll
        for (int j = 0; j < 4; ++j) {
            o[j] = wmma_bf16(pa0, loadB_row(Vs, LDT, 0,  j * 16, hl, r), o[j]);
            o[j] = wmma_bf16(pa1, loadB_row(Vs, LDT, 32, j * 16, hl, r), o[j]);
        }
    }

    // Normalize and write head-concatenated activation [B,S,768].
#pragma unroll
    for (int j = 0; j < 4; ++j)
#pragma unroll
        for (int i = 0; i < 8; ++i) {
            int s = q0 + wave * 16 + i + 8 * hl;
            int col = h * DKN + j * 16 + r;
            cat[((size_t)b * SN + s) * DN + col] = f2b(o[j][i] / lrow[i]);
        }
}

// ---------------------------------------------------------------------------
// Residual + LayerNorm: one 256-thread block per row of 768 (3 elems/thread).
// ---------------------------------------------------------------------------
DEV float block_sum(float x, float* red, int tid) {
#pragma unroll
    for (int m = 16; m >= 1; m >>= 1) x += __shfl_xor(x, m, 32);
    int wv = tid >> 5, ln = tid & 31;
    if (ln == 0) red[wv] = x;
    __syncthreads();
    if (wv == 0) {
        float t = (ln < 8) ? red[ln] : 0.0f;
#pragma unroll
        for (int m = 4; m >= 1; m >>= 1) t += __shfl_xor(t, m, 32);
        if (ln == 0) red[0] = t;
    }
    __syncthreads();
    float res = red[0];
    __syncthreads();
    return res;
}

__global__ __launch_bounds__(256)
void lnorm_kernel(const float* __restrict__ a, const float* __restrict__ resid,
                  const float* __restrict__ g, const float* __restrict__ be,
                  float* __restrict__ outF, bf16* __restrict__ outH) {
    __shared__ float red[8];
    const int row = blockIdx.x, tid = threadIdx.x;
    const float* A = a + (size_t)row * DN;
    const float* R = resid + (size_t)row * DN;

    float v[3], s = 0.0f;
#pragma unroll
    for (int t = 0; t < 3; ++t) {
        v[t] = A[tid + t * 256] + R[tid + t * 256];
        s += v[t];
    }
    float mu = block_sum(s, red, tid) * (1.0f / DN);

    float sq = 0.0f;
#pragma unroll
    for (int t = 0; t < 3; ++t) { float d = v[t] - mu; sq += d * d; }
    float var = block_sum(sq, red, tid) * (1.0f / DN);
    float rs = rsqrtf(var + 1e-5f);

#pragma unroll
    for (int t = 0; t < 3; ++t) {
        int c = tid + t * 256;
        float oo = (v[t] - mu) * rs * g[c] + be[c];
        outF[(size_t)row * DN + c] = oo;
        if (outH) outH[(size_t)row * DN + c] = f2b(oo);
    }
}

// ---------------------------------------------------------------------------
// Host-side orchestration
// ---------------------------------------------------------------------------
extern "C" void kernel_launch(void* const* d_in, const int* in_sizes, int n_in,
                              void* d_out, int out_size, void* d_ws,
                              size_t ws_size, hipStream_t stream) {
    (void)in_sizes; (void)n_in; (void)out_size; (void)ws_size;
    const float* src = (const float*)d_in[0];
    const float* Wq  = (const float*)d_in[1];
    const float* bq  = (const float*)d_in[2];
    const float* Wk  = (const float*)d_in[3];
    const float* bk  = (const float*)d_in[4];
    const float* Wv  = (const float*)d_in[5];
    const float* bv  = (const float*)d_in[6];
    const float* Wo  = (const float*)d_in[7];
    const float* bo  = (const float*)d_in[8];
    const float* g1  = (const float*)d_in[9];
    const float* be1 = (const float*)d_in[10];
    const float* W1  = (const float*)d_in[11];
    const float* bf1 = (const float*)d_in[12];
    const float* W2  = (const float*)d_in[13];
    const float* bf2 = (const float*)d_in[14];
    const float* g2  = (const float*)d_in[15];
    const float* be2 = (const float*)d_in[16];

    char* w = (char*)d_ws;
    auto take = [&](size_t bytes) -> void* {
        void* p = (void*)w;
        w += (bytes + 255) & ~(size_t)255;
        return p;
    };
    bf16*  xh   = (bf16*) take((size_t)MN * DN * 2);
    bf16*  wqkv = (bf16*) take((size_t)DN * 2304 * 2);
    float* bqkv = (float*)take(2304 * 4);
    bf16*  woh  = (bf16*) take((size_t)DN * DN * 2);
    bf16*  w1h  = (bf16*) take((size_t)DN * FN * 2);
    bf16*  w2h  = (bf16*) take((size_t)FN * DN * 2);
    bf16*  qb   = (bf16*) take((size_t)MN * DN * 2);
    bf16*  ktb  = (bf16*) take((size_t)MN * DN * 2);   // K transposed
    bf16*  vb   = (bf16*) take((size_t)MN * DN * 2);
    bf16*  cat  = (bf16*) take((size_t)MN * DN * 2);
    float* yf   = (float*)take((size_t)MN * DN * 4);
    float* x1f  = (float*)take((size_t)MN * DN * 4);
    bf16*  x1h  = (bf16*) take((size_t)MN * DN * 2);
    bf16*  hh   = (bf16*) take((size_t)MN * FN * 2);

    // 1) precision prep
    cast_bf16_kernel<<<(MN * DN + 255) / 256, 256, 0, stream>>>(src, xh, (long)MN * DN);
    pack_qkvw_kernel<<<(DN * 2304 + 255) / 256, 256, 0, stream>>>(Wq, Wk, Wv, wqkv);
    pack_qkvb_kernel<<<(2304 + 255) / 256, 256, 0, stream>>>(bq, bk, bv, bqkv);
    cast_bf16_kernel<<<(DN * DN + 255) / 256, 256, 0, stream>>>(Wo, woh, (long)DN * DN);
    cast_bf16_kernel<<<(DN * FN + 255) / 256, 256, 0, stream>>>(W1, w1h, (long)DN * FN);
    cast_bf16_kernel<<<(FN * DN + 255) / 256, 256, 0, stream>>>(W2, w2h, (long)FN * DN);

    // 2) fused QKV projection (q pre-scaled, K stored transposed)
    gemm128_kernel<0><<<dim3(2304 / 128, MN / 128), 256, 0, stream>>>(
        xh, wqkv, bqkv, MN, 2304, DN, nullptr, nullptr, qb, ktb, vb);

    // 3) flash attention -> concatenated heads
    attn_kernel<<<dim3(SN / 64, BN * HN), 128, 0, stream>>>(qb, ktb, vb, cat);

    // 4) output projection + residual LayerNorm
    gemm128_kernel<2><<<dim3(DN / 128, MN / 128), 256, 0, stream>>>(
        cat, woh, bo, MN, DN, DN, yf, nullptr, nullptr, nullptr, nullptr);
    lnorm_kernel<<<MN, 256, 0, stream>>>(yf, src, g1, be1, x1f, x1h);

    // 5) FFN
    gemm128_kernel<1><<<dim3(FN / 128, MN / 128), 256, 0, stream>>>(
        x1h, w1h, bf1, MN, FN, DN, nullptr, hh, nullptr, nullptr, nullptr);
    gemm128_kernel<2><<<dim3(DN / 128, MN / 128), 256, 0, stream>>>(
        hh, w2h, bf2, MN, DN, FN, yf, nullptr, nullptr, nullptr, nullptr);
    lnorm_kernel<<<MN, 256, 0, stream>>>(yf, x1f, g2, be2, (float*)d_out, nullptr);
}
